// RTSFNetANN_25220047962147
// MI455X (gfx1250) — matrix operand order repaired and checked by reading, not compiled, not run on hardware
//
#include <hip/hip_runtime.h>
#include <hip/hip_bf16.h>
#include <math.h>

// ---------------------------------------------------------------------------
// Problem constants (from reference)
// ---------------------------------------------------------------------------
#define BATCH 64
#define TLEN  4096
#define INF_  48      // C*V = 3*16
#define HEADS 4
#define ROT_H 128
#define MLP_H 256
#define NCLS  12
#define FDIM  84      // HEADS*3*7

#if defined(__HIP_DEVICE_COMPILE__) && defined(__gfx1250__)
#define ASYNC_LDS 1
#endif

typedef __attribute__((ext_vector_type(16))) _Float16 v16h;
typedef __attribute__((ext_vector_type(8)))  float    v8f;

// ---------------------------------------------------------------------------
// WMMA fragment loaders (CDNA5 16x16x32 f16 layouts, ISA 7.12.2, wave32)
// ---------------------------------------------------------------------------
__device__ __forceinline__ v16h load_a_f32src(const float* X, int ldx, int m0,
                                              int kk, int kdim, int lane) {
  const int hs = (lane >= 16) ? 1 : 0;
  const int m  = m0 + (lane & 15);
  v16h a;
#pragma unroll
  for (int e = 0; e < 16; ++e) {
    int k = kk + (e & 7) + ((e >> 3) << 4) + hs * 8;
    float v = (k < kdim) ? X[m * ldx + k] : 0.0f;
    a[e] = (_Float16)v;
  }
  return a;
}

__device__ __forceinline__ v16h load_a_f16src(const _Float16* X, int ldx, int m0,
                                              int kk, int lane) {
  const int hs = (lane >= 16) ? 1 : 0;
  const int m  = m0 + (lane & 15);
  v16h a;
#pragma unroll
  for (int e = 0; e < 16; ++e) {
    int k = kk + (e & 7) + ((e >> 3) << 4) + hs * 8;
    a[e] = X[m * ldx + k];
  }
  return a;
}

__device__ __forceinline__ v16h load_b_f32src(const float* W, int ldw, int n0,
                                              int kk, int kdim, int ndim, int lane) {
  const int hs = (lane >= 16) ? 1 : 0;
  const int n  = n0 + (lane & 15);
  v16h b;
#pragma unroll
  for (int e = 0; e < 16; ++e) {
    int k = kk + (e & 7) + ((e >> 3) << 4) + hs * 8;
    float v = (k < kdim && n < ndim) ? W[k * ldw + n] : 0.0f;
    b[e] = (_Float16)v;
  }
  return b;
}

// ---------------------------------------------------------------------------
// Kernel 0: zero the feature accumulator
// ---------------------------------------------------------------------------
__global__ void k_zero(float* __restrict__ p, int n) {
  int i = blockIdx.x * 256 + threadIdx.x;
  if (i < n) p[i] = 0.0f;
}

// ---------------------------------------------------------------------------
// Kernel 1: gf[b][f] = mean_t x[b][t][f]   (pass 1, b128 loads)
// grid: BATCH blocks, 256 threads. Each thread reads 192 float4s; the
// feature base of float4 #k is (4*tid + 16*k) mod 48 -> period-3 phases.
// ---------------------------------------------------------------------------
__global__ void __launch_bounds__(256) k_colmean(const float* __restrict__ x,
                                                 float* __restrict__ gf) {
  const int b = blockIdx.x;
  const int tid = threadIdx.x;
  const float4* xb4 = (const float4*)(x + (size_t)b * TLEN * INF_);
  float acc[3][4];
#pragma unroll
  for (int p = 0; p < 3; ++p)
#pragma unroll
    for (int j = 0; j < 4; ++j) acc[p][j] = 0.0f;

  // TLEN*INF_/4 = 49152 float4 per batch -> 192 per thread, unrolled by 3
  for (int k0 = 0; k0 < 192; k0 += 3) {
#pragma unroll
    for (int p = 0; p < 3; ++p) {
      float4 v = xb4[tid + 256 * (k0 + p)];
      acc[p][0] += v.x; acc[p][1] += v.y; acc[p][2] += v.z; acc[p][3] += v.w;
    }
  }

  __shared__ float red[INF_];
  if (tid < INF_) red[tid] = 0.0f;
  __syncthreads();
#pragma unroll
  for (int p = 0; p < 3; ++p) {
    const int fb = (4 * tid + 16 * p) % 48;
#pragma unroll
    for (int j = 0; j < 4; ++j) atomicAdd(&red[fb + j], acc[p][j]);
  }
  __syncthreads();
  if (tid < INF_) gf[b * INF_ + tid] = red[tid] * (1.0f / (float)TLEN);
}

// ---------------------------------------------------------------------------
// Kernel 2: rotation MLP (gf -> h1 -> h2 -> rot) + Rodrigues coefficients
// single workgroup, 256 threads = 8 waves, WMMA f16 GEMMs
// ---------------------------------------------------------------------------
__global__ void __launch_bounds__(256) k_rotmlp(
    const float* __restrict__ gf,
    const float* __restrict__ w1, const float* __restrict__ b1,
    const float* __restrict__ w2, const float* __restrict__ b2,
    const float* __restrict__ w3, const float* __restrict__ b3,
    float* __restrict__ coef) {
  __shared__ _Float16 h1[BATCH * ROT_H];   // 16 KB
  __shared__ _Float16 h2[BATCH * ROT_H];   // 16 KB
  __shared__ float    rb[BATCH * 16];      // 4 KB
  const int tid = threadIdx.x;
  const int lane = tid & 31;
  const int w = tid >> 5;
  const int hs = (lane >= 16) ? 1 : 0;
  const int nl = lane & 15;

  // stage 1: h1 = relu(gf[64x48] @ w1[48x128] + b1)
  for (int tile = w; tile < 32; tile += 8) {
    const int m0 = (tile >> 3) * 16, n0 = (tile & 7) * 16;
    v8f acc = {};
    for (int kk = 0; kk < 64; kk += 32) {
      v16h a = load_a_f32src(gf, INF_, m0, kk, INF_, lane);
      v16h b = load_b_f32src(w1, ROT_H, n0, kk, INF_, ROT_H, lane);
      acc = __builtin_amdgcn_wmma_f32_16x16x32_f16(false, a, false, b,
                                                   (short)0, acc, false, false);
    }
    const int n = n0 + nl;
    const float bias = b1[n];
#pragma unroll
    for (int r = 0; r < 8; ++r) {
      const int m = m0 + r + hs * 8;
      float v = acc[r] + bias;
      h1[m * ROT_H + n] = (_Float16)(v > 0.0f ? v : 0.0f);
    }
  }
  __syncthreads();

  // stage 2: h2 = relu(h1 @ w2[128x128] + b2)
  for (int tile = w; tile < 32; tile += 8) {
    const int m0 = (tile >> 3) * 16, n0 = (tile & 7) * 16;
    v8f acc = {};
    for (int kk = 0; kk < ROT_H; kk += 32) {
      v16h a = load_a_f16src(h1, ROT_H, m0, kk, lane);
      v16h b = load_b_f32src(w2, ROT_H, n0, kk, ROT_H, ROT_H, lane);
      acc = __builtin_amdgcn_wmma_f32_16x16x32_f16(false, a, false, b,
                                                   (short)0, acc, false, false);
    }
    const int n = n0 + nl;
    const float bias = b2[n];
#pragma unroll
    for (int r = 0; r < 8; ++r) {
      const int m = m0 + r + hs * 8;
      float v = acc[r] + bias;
      h2[m * ROT_H + n] = (_Float16)(v > 0.0f ? v : 0.0f);
    }
  }
  __syncthreads();

  // stage 3: rb = h2 @ w3[128x16] + b3
  if (w < 4) {
    const int m0 = w * 16;
    v8f acc = {};
    for (int kk = 0; kk < ROT_H; kk += 32) {
      v16h a = load_a_f16src(h2, ROT_H, m0, kk, lane);
      v16h b = load_b_f32src(w3, 16, 0, kk, ROT_H, 16, lane);
      acc = __builtin_amdgcn_wmma_f32_16x16x32_f16(false, a, false, b,
                                                   (short)0, acc, false, false);
    }
    const int n = nl;
    const float bias = b3[n];
#pragma unroll
    for (int r = 0; r < 8; ++r) {
      const int m = m0 + r + hs * 8;
      rb[m * 16 + n] = acc[r] + bias;
    }
  }
  __syncthreads();

  // Rodrigues coefficients: one thread per (b, head)
  {
    const int b = tid >> 2;
    const int hh = tid & 3;
    const float r0 = tanhf(rb[b * 16 + hh * 4 + 0]);
    const float r1 = tanhf(rb[b * 16 + hh * 4 + 1]);
    const float r2 = tanhf(rb[b * 16 + hh * 4 + 2]);
    const float r3 = tanhf(rb[b * 16 + hh * 4 + 3]);
    const float nrm = sqrtf(r0 * r0 + r1 * r1 + r2 * r2) + 1e-8f;
    const float kx = r0 / nrm, ky = r1 / nrm, kz = r2 / nrm;
    const float th = r3 * 3.14159265358979323846f;
    const float ct = cosf(th), st = sinf(th);
    float* cp = coef + (b * HEADS + hh) * 8;
    cp[0] = kx; cp[1] = ky; cp[2] = kz;
    cp[3] = ct; cp[4] = st; cp[5] = 1.0f - ct;
  }
}

// ---------------------------------------------------------------------------
// Kernel 3: fused rotation + hierarchical block statistics (pass 2)
// grid: (16 chunks, BATCH), block 256: tid -> g(16) x h(4) x q(4)
// The 48KB x-chunk is staged into LDS via global_load_async_to_lds_b128
// (ASYNCcnt DMA path, 4x reuse by the 4 head groups), then each thread
// streams one (b,h,64-t block,xyz group) out of LDS.
// ---------------------------------------------------------------------------
#define STAT_RESET(S) do {                                   \
    sum[S][c] = 0.0f; sq[S][c] = 0.0f; mac[S][c] = 0.0f;     \
    mn[S][c] = 3.402823466e38f; mx[S][c] = -3.402823466e38f; \
  } while (0)

#define EMIT(S, BS) do {                                     \
    _Pragma("unroll")                                        \
    for (int c = 0; c < 3; ++c) {                            \
      const float inv = 1.0f / (float)(BS);                  \
      float mean = sum[S][c] * inv;                          \
      float e2   = sq[S][c] * inv;                           \
      float var  = e2 - mean * mean;                         \
      var = var > 0.0f ? var : 0.0f;                         \
      tot[(S)*7 + 0] += mean;                                \
      tot[(S)*7 + 1] += sqrtf(var);                          \
      tot[(S)*7 + 2] += mn[S][c];                            \
      tot[(S)*7 + 3] += mx[S][c];                            \
      tot[(S)*7 + 4] += sqrtf(e2 + 1e-8f);                   \
      tot[(S)*7 + 5] += mac[S][c] * (1.0f / (float)((BS)-1));\
      tot[(S)*7 + 6] += e2;                                  \
      STAT_RESET(S);                                         \
    }                                                        \
  } while (0)

__global__ void __launch_bounds__(256) k_rotstats(const float* __restrict__ x,
                                                  const float* __restrict__ coef,
                                                  float* __restrict__ featacc) {
  const int b = blockIdx.y;
  const int chunk = blockIdx.x;       // 0..15, 256 timesteps each
  const int tid = threadIdx.x;
  const int g = tid & 15;
  const int h = (tid >> 4) & 3;
  const int q = tid >> 6;             // 0..3 -> 64-t block within chunk

  __shared__ __align__(16) float xs[256 * INF_];   // 48 KB staged chunk
  __shared__ float lred[HEADS * 21];

  // ---- stage x[b][chunk*256 .. +255][0..47] into LDS -----------------------
  {
    const float* gsrc = x + ((size_t)b * TLEN + (size_t)chunk * 256) * INF_;
#ifdef ASYNC_LDS
    // low 32 bits of a generic pointer to LDS == wave-relative LDS offset
    const unsigned lbase = (unsigned)(unsigned long long)(const void*)xs;
#pragma unroll
    for (int k = 0; k < 12; ++k) {
      const int idx = tid + 256 * k;                 // float4 index
      unsigned loff = lbase + (unsigned)idx * 16u;
      unsigned long long ga =
          (unsigned long long)(const void*)(gsrc + (size_t)idx * 4);
      asm volatile("global_load_async_to_lds_b128 %0, %1, off"
                   :: "v"(loff), "v"(ga) : "memory");
    }
    asm volatile("s_wait_asynccnt 0x0" ::: "memory");
#else
    for (int k = 0; k < 12; ++k) {
      const int idx = tid + 256 * k;
      ((float4*)xs)[idx] = ((const float4*)gsrc)[idx];
    }
#endif
  }
  __syncthreads();

  // ---- rotation coefficients ----------------------------------------------
  const float* cp = coef + (b * HEADS + h) * 8;
  const float kx = cp[0], ky = cp[1], kz = cp[2];
  const float ct = cp[3], st = cp[4], omc = cp[5];

  float sum[3][3], sq[3][3], mn[3][3], mx[3][3], mac[3][3], prev[3];
  float tot[21];
#pragma unroll
  for (int s = 0; s < 3; ++s)
#pragma unroll
    for (int c = 0; c < 3; ++c) { sum[s][c] = 0.f; sq[s][c] = 0.f; mac[s][c] = 0.f;
                                  mn[s][c] = 3.402823466e38f; mx[s][c] = -3.402823466e38f; }
#pragma unroll
  for (int j = 0; j < 21; ++j) tot[j] = 0.0f;
  prev[0] = prev[1] = prev[2] = 0.0f;

  for (int blk = 0; blk < 4; ++blk) {         // four 16-blocks of the 64-block
#pragma unroll
    for (int ii = 0; ii < 16; ++ii) {
      const int lt = q * 64 + blk * 16 + ii;  // local timestep in chunk
      const float* p = xs + lt * INF_ + g * 3;
      const float v0 = p[0], v1 = p[1], v2 = p[2];
      const float cx = ky * v2 - kz * v1;
      const float cy = kz * v0 - kx * v2;
      const float cz = kx * v1 - ky * v0;
      const float dot = kx * v0 + ky * v1 + kz * v2;
      float z[3];
      z[0] = v0 * ct + cx * st + kx * dot * omc;
      z[1] = v1 * ct + cy * st + ky * dot * omc;
      z[2] = v2 * ct + cz * st + kz * dot * omc;
      const bool notFirst = (ii != 0) || (blk != 0);      // i > 0
      const bool in16 = (ii != 0);                        // i % 16 != 0
#pragma unroll
      for (int c = 0; c < 3; ++c) {
        const float zz = z[c];
        if (notFirst) {
          const float d = fabsf(zz - prev[c]);
          mac[2][c] += d;                                  // 64-block interior
          if (in16 || (blk & 1)) mac[1][c] += d;           // i % 32 != 0
          if (in16)              mac[0][c] += d;           // i % 16 != 0
        }
        prev[c] = zz;
#pragma unroll
        for (int s = 0; s < 3; ++s) {
          sum[s][c] += zz;
          sq[s][c]  = fmaf(zz, zz, sq[s][c]);
          mn[s][c]  = fminf(mn[s][c], zz);
          mx[s][c]  = fmaxf(mx[s][c], zz);
        }
      }
    }
    EMIT(0, 16);
    if (blk & 1) EMIT(1, 32);
  }
  EMIT(2, 64);

  // ---- workgroup reduction: LDS f32 atomics, then global f32 atomics ------
  for (int j = tid; j < HEADS * 21; j += 256) lred[j] = 0.0f;
  __syncthreads();
#pragma unroll
  for (int j = 0; j < 21; ++j) atomicAdd(&lred[h * 21 + j], tot[j]);
  __syncthreads();
  if (tid < HEADS * 21) atomicAdd(&featacc[b * FDIM + tid], lred[tid]);
}

// ---------------------------------------------------------------------------
// Kernel 4: final MLP  feat[64x84] -> 256 -> 256 -> 12  (WMMA, 1 workgroup)
// ---------------------------------------------------------------------------
__global__ void __launch_bounds__(256) k_mlp(
    const float* __restrict__ featacc,
    const float* __restrict__ p1, const float* __restrict__ pb1,
    const float* __restrict__ p2, const float* __restrict__ pb2,
    const float* __restrict__ p3, const float* __restrict__ pb3,
    float* __restrict__ out) {
  __shared__ _Float16 bufA[BATCH * MLP_H];   // 32 KB: feat (64x96) then q2
  __shared__ _Float16 bufB[BATCH * MLP_H];   // 32 KB: q1
  const int tid = threadIdx.x;
  const int lane = tid & 31;
  const int w = tid >> 5;
  const int hs = (lane >= 16) ? 1 : 0;
  const int nl = lane & 15;

  // normalize feature accumulator into f16 feat[64][96] (K padded 84 -> 96)
  for (int idx = tid; idx < BATCH * 96; idx += 256) {
    const int b = idx / 96, j = idx % 96;
    float v = 0.0f;
    if (j < FDIM) {
      const int s = (j % 21) / 7;
      const float nblk = (s == 0) ? 256.0f : (s == 1) ? 128.0f : 64.0f; // T/bs
      v = featacc[b * FDIM + j] * (1.0f / (nblk * 48.0f));
    }
    bufA[idx] = (_Float16)v;
  }
  __syncthreads();

  // stage 1: q1 = relu(feat @ p1[84x256] + pb1)
  for (int tile = w; tile < 64; tile += 8) {
    const int m0 = (tile >> 4) * 16, n0 = (tile & 15) * 16;
    v8f acc = {};
    for (int kk = 0; kk < 96; kk += 32) {
      v16h a = load_a_f16src(bufA, 96, m0, kk, lane);
      v16h b = load_b_f32src(p1, MLP_H, n0, kk, FDIM, MLP_H, lane);
      acc = __builtin_amdgcn_wmma_f32_16x16x32_f16(false, a, false, b,
                                                   (short)0, acc, false, false);
    }
    const int n = n0 + nl;
    const float bias = pb1[n];
#pragma unroll
    for (int r = 0; r < 8; ++r) {
      const int m = m0 + r + hs * 8;
      float v = acc[r] + bias;
      bufB[m * MLP_H + n] = (_Float16)(v > 0.0f ? v : 0.0f);
    }
  }
  __syncthreads();

  // stage 2: q2 = relu(q1 @ p2[256x256] + pb2) -> bufA
  for (int tile = w; tile < 64; tile += 8) {
    const int m0 = (tile >> 4) * 16, n0 = (tile & 15) * 16;
    v8f acc = {};
    for (int kk = 0; kk < MLP_H; kk += 32) {
      v16h a = load_a_f16src(bufB, MLP_H, m0, kk, lane);
      v16h b = load_b_f32src(p2, MLP_H, n0, kk, MLP_H, MLP_H, lane);
      acc = __builtin_amdgcn_wmma_f32_16x16x32_f16(false, a, false, b,
                                                   (short)0, acc, false, false);
    }
    const int n = n0 + nl;
    const float bias = pb2[n];
#pragma unroll
    for (int r = 0; r < 8; ++r) {
      const int m = m0 + r + hs * 8;
      float v = acc[r] + bias;
      bufA[m * MLP_H + n] = (_Float16)(v > 0.0f ? v : 0.0f);
    }
  }
  __syncthreads();

  // stage 3: out = q2 @ p3[256x12] + pb3
  if (w < 4) {
    const int m0 = w * 16;
    v8f acc = {};
    for (int kk = 0; kk < MLP_H; kk += 32) {
      v16h a = load_a_f16src(bufA, MLP_H, m0, kk, lane);
      v16h b = load_b_f32src(p3, NCLS, 0, kk, MLP_H, NCLS, lane);
      acc = __builtin_amdgcn_wmma_f32_16x16x32_f16(false, a, false, b,
                                                   (short)0, acc, false, false);
    }
    const int n = nl;
    if (n < NCLS) {
      const float bias = pb3[n];
#pragma unroll
      for (int r = 0; r < 8; ++r) {
        const int m = m0 + r + hs * 8;
        out[m * NCLS + n] = acc[r] + bias;
      }
    }
  }
}

// ---------------------------------------------------------------------------
// Host launcher
// ---------------------------------------------------------------------------
extern "C" void kernel_launch(void* const* d_in, const int* in_sizes, int n_in,
                              void* d_out, int out_size, void* d_ws, size_t ws_size,
                              hipStream_t stream) {
  const float* x  = (const float*)d_in[0];
  const float* w1 = (const float*)d_in[1];
  const float* b1 = (const float*)d_in[2];
  const float* w2 = (const float*)d_in[3];
  const float* b2 = (const float*)d_in[4];
  const float* w3 = (const float*)d_in[5];
  const float* b3 = (const float*)d_in[6];
  const float* p1 = (const float*)d_in[7];
  const float* pb1= (const float*)d_in[8];
  const float* p2 = (const float*)d_in[9];
  const float* pb2= (const float*)d_in[10];
  const float* p3 = (const float*)d_in[11];
  const float* pb3= (const float*)d_in[12];

  float* ws      = (float*)d_ws;
  float* gf      = ws;                         // 64*48   = 3072
  float* coef    = ws + 3072;                  // 64*4*8  = 2048
  float* featacc = ws + 3072 + 2048;           // 64*84   = 5376

  k_zero<<<(BATCH * FDIM + 255) / 256, 256, 0, stream>>>(featacc, BATCH * FDIM);
  k_colmean<<<BATCH, 256, 0, stream>>>(x, gf);
  k_rotmlp<<<1, 256, 0, stream>>>(gf, w1, b1, w2, b2, w3, b3, coef);
  k_rotstats<<<dim3(16, BATCH), 256, 0, stream>>>(x, coef, featacc);
  k_mlp<<<1, 256, 0, stream>>>(featacc, p1, pb1, p2, pb2, p3, pb3, (float*)d_out);
}
